// LSTMTextEncoder_37684043055273
// MI455X (gfx1250) — matrix operand order repaired and checked by reading, not compiled, not run on hardware
//
#include <hip/hip_runtime.h>

// ---------------- types ----------------
typedef __bf16 bf16;
typedef __attribute__((ext_vector_type(16))) __bf16 v16bf;
typedef __attribute__((ext_vector_type(8)))  float  v8f;

struct alignas(16) U4 { unsigned int x, y, z, w; };
union Frag { v16bf v; U4 u[2]; };

constexpr int kE = 512;     // embedding dim
constexpr int kH = 512;     // hidden dim
constexpr int kB = 256;     // batch
constexpr int kT = 128;     // seq len
constexpr int kK = 1024;    // fused K = E + H

// =====================================================================
// Setup kernel 1: pack fused gate weight W^T[n][k], n in [0,2048), k in
// [0,1024): rows 0..511 of K come from Wx[E,4H], 512..1023 from Uh[H,4H].
// Stored transposed+bf16 so B fragments are 32 contiguous bytes per lane.
// =====================================================================
__global__ __launch_bounds__(256) void pack_gate_w(
    const float* __restrict__ Wx, const float* __restrict__ Uh,
    bf16* __restrict__ Wt)
{
    int idx = blockIdx.x * 256 + threadIdx.x;     // 2048*1024 threads
    int n = idx >> 10;
    int k = idx & 1023;
    float v = (k < kE) ? Wx[(size_t)k * 2048 + n]
                       : Uh[(size_t)(k - kE) * 2048 + n];
    Wt[idx] = (bf16)v;                            // Wt[n*1024 + k]
}

// Setup kernel 2: pack fcW^T[n][k] bf16 (n in [0,512) = E, k in [0,512) = H)
__global__ __launch_bounds__(256) void pack_fc_w(
    const float* __restrict__ fcW, bf16* __restrict__ fcWt)
{
    int idx = blockIdx.x * 256 + threadIdx.x;     // 512*512 threads
    int n = idx >> 9;
    int k = idx & 511;
    fcWt[idx] = (bf16)fcW[(size_t)k * kE + n];
}

// Setup kernel 3: combined gate bias (bx+bu), zero h0 and c
__global__ __launch_bounds__(256) void init_state(
    const float* __restrict__ bx, const float* __restrict__ bu,
    float* __restrict__ bias, bf16* __restrict__ h0, float* __restrict__ c)
{
    int idx = blockIdx.x * 256 + threadIdx.x;     // 256*512 threads
    if (idx < 2048) bias[idx] = bx[idx] + bu[idx];
    h0[idx] = (bf16)0.0f;
    c[idx]  = 0.0f;
}

// =====================================================================
// Gather embeddings -> xseq[t][b][e] bf16 (padding_idx==0 -> zero row)
// =====================================================================
__global__ __launch_bounds__(256) void gather_embed(
    const int* __restrict__ captions, const float* __restrict__ emb,
    bf16* __restrict__ xseq)
{
    int idx = blockIdx.x * 256 + threadIdx.x;     // B*T*(E/8) = 2,097,152
    int e8  = idx & 63;
    int rem = idx >> 6;
    int t   = rem & 127;
    int b   = rem >> 7;
    int tok = captions[b * kT + t];
    float4 v0 = make_float4(0.f, 0.f, 0.f, 0.f);
    float4 v1 = v0;
    if (tok != 0) {
        const float4* src = (const float4*)(emb + ((size_t)tok * kE + e8 * 8));
        v0 = src[0];
        v1 = src[1];
    }
    union { bf16 h[8]; U4 u; } pk;
    pk.h[0] = (bf16)v0.x; pk.h[1] = (bf16)v0.y;
    pk.h[2] = (bf16)v0.z; pk.h[3] = (bf16)v0.w;
    pk.h[4] = (bf16)v1.x; pk.h[5] = (bf16)v1.y;
    pk.h[6] = (bf16)v1.z; pk.h[7] = (bf16)v1.w;
    *(U4*)(xseq + ((size_t)(t * kB + b) * kE + e8 * 8)) = pk.u;
}

// ---------------- WMMA helpers ----------------
__device__ __forceinline__ v16bf load_a_frag(const bf16* asrc, int lane_hi) {
    // A 16x32 bf16 layout: lanes 0-15 -> K 0..15, lanes 16-31 -> K 16..31 (contiguous)
    Frag a;
    const U4* ap = (const U4*)(asrc + lane_hi * 16);
    a.u[0] = ap[0];
    a.u[1] = ap[1];   // +8 elements
    return a.v;
}
__device__ __forceinline__ v16bf load_b_frag(const bf16* brow, int lane_hi) {
    // B 32x16 from W^T row: lanes 0-15 -> K 0..15, lanes 16-31 -> K 16..31 (contiguous)
    Frag b;
    const U4* bp = (const U4*)(brow + lane_hi * 16);
    b.u[0] = bp[0];
    b.u[1] = bp[1];
    return b.v;
}
__device__ __forceinline__ float sigmoidf_(float x) {
    return 1.0f / (1.0f + __expf(-x));
}

// =====================================================================
// One LSTM time step: g = [x_t ; h] @ [Wx;Uh]^T-packed + bias, then cell.
// Grid: 128 WGs x 128 thr (4 waves). Each wave owns a 16(M)x16(N) slice
// of ALL FOUR gates -> in-register cell update, c in place, h ping-pong.
// =====================================================================
__global__ __launch_bounds__(128) void lstm_step(
    const bf16* __restrict__ xt,    // [256][512]  x_t bf16
    const bf16* __restrict__ h_in,  // [256][512]
    bf16*       __restrict__ h_out, // [256][512]
    const bf16* __restrict__ Wt,    // [2048][1024] packed W^T bf16
    const float* __restrict__ bias, // [2048] bx+bu
    float*      __restrict__ c)     // [256][512] cell state (in place)
{
    const int lane    = threadIdx.x & 31;
    const int wave    = threadIdx.x >> 5;
    const int lane_lo = lane & 15;
    const int lane_hi = lane >> 4;
    const int mb = blockIdx.x >> 3;                 // 0..15  (M block)
    const int j0 = (blockIdx.x & 7) * 64 + wave * 16; // 0..496 (per-gate col)
    const int m0 = mb * 16;

    const bf16* arow_x = xt   + (size_t)(m0 + lane_lo) * kE;
    const bf16* arow_h = h_in + (size_t)(m0 + lane_lo) * kH;
    const bf16* br0 = Wt + (size_t)(0 * kH + j0 + lane_lo) * kK;
    const bf16* br1 = Wt + (size_t)(1 * kH + j0 + lane_lo) * kK;
    const bf16* br2 = Wt + (size_t)(2 * kH + j0 + lane_lo) * kK;
    const bf16* br3 = Wt + (size_t)(3 * kH + j0 + lane_lo) * kK;

    v8f acc0 = {0,0,0,0,0,0,0,0};
    v8f acc1 = {0,0,0,0,0,0,0,0};
    v8f acc2 = {0,0,0,0,0,0,0,0};
    v8f acc3 = {0,0,0,0,0,0,0,0};

    // K = 0..511 : input projection (x_t part)
    for (int kb = 0; kb < 16; ++kb) {
        v16bf a  = load_a_frag(arow_x + kb * 32, lane_hi);
        v16bf b0 = load_b_frag(br0 + kb * 32, lane_hi);
        v16bf b1 = load_b_frag(br1 + kb * 32, lane_hi);
        v16bf b2 = load_b_frag(br2 + kb * 32, lane_hi);
        v16bf b3 = load_b_frag(br3 + kb * 32, lane_hi);
        acc0 = __builtin_amdgcn_wmma_f32_16x16x32_bf16(false, a, false, b0, (short)0, acc0, false, false);
        acc1 = __builtin_amdgcn_wmma_f32_16x16x32_bf16(false, a, false, b1, (short)0, acc1, false, false);
        acc2 = __builtin_amdgcn_wmma_f32_16x16x32_bf16(false, a, false, b2, (short)0, acc2, false, false);
        acc3 = __builtin_amdgcn_wmma_f32_16x16x32_bf16(false, a, false, b3, (short)0, acc3, false, false);
    }
    // K = 512..1023 : recurrent projection (h part)
    for (int kb = 16; kb < 32; ++kb) {
        v16bf a  = load_a_frag(arow_h + (kb - 16) * 32, lane_hi);
        v16bf b0 = load_b_frag(br0 + kb * 32, lane_hi);
        v16bf b1 = load_b_frag(br1 + kb * 32, lane_hi);
        v16bf b2 = load_b_frag(br2 + kb * 32, lane_hi);
        v16bf b3 = load_b_frag(br3 + kb * 32, lane_hi);
        acc0 = __builtin_amdgcn_wmma_f32_16x16x32_bf16(false, a, false, b0, (short)0, acc0, false, false);
        acc1 = __builtin_amdgcn_wmma_f32_16x16x32_bf16(false, a, false, b1, (short)0, acc1, false, false);
        acc2 = __builtin_amdgcn_wmma_f32_16x16x32_bf16(false, a, false, b2, (short)0, acc2, false, false);
        acc3 = __builtin_amdgcn_wmma_f32_16x16x32_bf16(false, a, false, b3, (short)0, acc3, false, false);
    }

    const int ncol = j0 + lane_lo;
    const float bi = bias[0 * kH + ncol];
    const float bf = bias[1 * kH + ncol];
    const float bo = bias[2 * kH + ncol];
    const float bg = bias[3 * kH + ncol];

    // C/D layout: VGPR r, lanes 0-15 -> M=r, lanes 16-31 -> M=8+r, N=lane&15
#pragma unroll
    for (int r = 0; r < 8; ++r) {
        const int m   = m0 + r + 8 * lane_hi;
        const int idx = m * kH + ncol;
        const float iv = sigmoidf_(acc0[r] + bi);
        const float fv = sigmoidf_(acc1[r] + bf);
        const float ov = sigmoidf_(acc2[r] + bo);
        const float gv = tanhf(acc3[r] + bg);
        const float cn = fv * c[idx] + iv * gv;
        c[idx] = cn;
        h_out[idx] = (bf16)(ov * tanhf(cn));
    }
}

// =====================================================================
// Final: out = h @ fcW + fcb, then row-wise L2 normalize (eps=1e-12).
// 16 WGs x 256 thr; WG owns 16 rows x all 512 cols -> LDS row reduction.
// =====================================================================
__global__ __launch_bounds__(256) void fc_normalize(
    const bf16* __restrict__ h,     // [256][512] bf16
    const bf16* __restrict__ fcWt,  // [512][512] packed fcW^T bf16
    const float* __restrict__ fcb,  // [512]
    float* __restrict__ out)        // [256][512] f32
{
    __shared__ float tile[16 * 520];
    __shared__ float rowsum[16];

    const int lane    = threadIdx.x & 31;
    const int wave    = threadIdx.x >> 5;
    const int lane_lo = lane & 15;
    const int lane_hi = lane >> 4;
    const int m0 = blockIdx.x * 16;

    const bf16* arow = h + (size_t)(m0 + lane_lo) * kH;

    v8f acc[4];
#pragma unroll
    for (int q = 0; q < 4; ++q) { v8f z = {0,0,0,0,0,0,0,0}; acc[q] = z; }

    for (int kb = 0; kb < 16; ++kb) {
        v16bf a = load_a_frag(arow + kb * 32, lane_hi);
#pragma unroll
        for (int q = 0; q < 4; ++q) {
            const int j0q = (wave + 8 * q) * 16;
            v16bf b = load_b_frag(fcWt + (size_t)(j0q + lane_lo) * kH + kb * 32, lane_hi);
            acc[q] = __builtin_amdgcn_wmma_f32_16x16x32_bf16(false, a, false, b, (short)0, acc[q], false, false);
        }
    }

#pragma unroll
    for (int q = 0; q < 4; ++q) {
        const int n = (wave + 8 * q) * 16 + lane_lo;
        const float bn = fcb[n];
#pragma unroll
        for (int r = 0; r < 8; ++r) {
            const int rr = r + 8 * lane_hi;
            tile[rr * 520 + n] = acc[q][r] + bn;
        }
    }
    __syncthreads();
    if (threadIdx.x < 16) rowsum[threadIdx.x] = 0.0f;
    __syncthreads();

    const int row = threadIdx.x >> 4;
    const int seg = threadIdx.x & 15;
    float s = 0.0f;
#pragma unroll
    for (int i = 0; i < 32; ++i) {
        const float v = tile[row * 520 + seg * 32 + i];
        s += v * v;
    }
    atomicAdd(&rowsum[row], s);
    __syncthreads();

    const float scale = 1.0f / fmaxf(sqrtf(rowsum[row]), 1e-12f);
#pragma unroll
    for (int i = 0; i < 32; ++i) {
        out[(size_t)(m0 + row) * kE + seg * 32 + i] =
            tile[row * 520 + seg * 32 + i] * scale;
    }
}

// =====================================================================
// host launcher
// =====================================================================
extern "C" void kernel_launch(void* const* d_in, const int* in_sizes, int n_in,
                              void* d_out, int out_size, void* d_ws, size_t ws_size,
                              hipStream_t stream) {
    (void)in_sizes; (void)n_in; (void)out_size; (void)ws_size;
    const int*   captions = (const int*)  d_in[0];
    const float* emb      = (const float*)d_in[1];
    const float* Wx       = (const float*)d_in[2];
    const float* bx       = (const float*)d_in[3];
    const float* Uh       = (const float*)d_in[4];
    const float* bu       = (const float*)d_in[5];
    const float* fcW      = (const float*)d_in[6];
    const float* fcb      = (const float*)d_in[7];

    char* ws = (char*)d_ws;
    bf16*  Wt   = (bf16*) (ws);                     //  4,194,304 B  [2048][1024] bf16
    bf16*  fcWt = (bf16*) (ws + 4194304);           //    524,288 B  [512][512]  bf16
    float* bias = (float*)(ws + 4718592);           //      8,192 B  [2048] f32
    bf16*  xseq = (bf16*) (ws + 4726784);           // 33,554,432 B  [128][256][512] bf16
    bf16*  h0   = (bf16*) (ws + 38281216);          //    262,144 B
    bf16*  h1   = (bf16*) (ws + 38543360);          //    262,144 B
    float* c    = (float*)(ws + 38805504);          //    524,288 B  (total ~37.5 MB)

    pack_gate_w<<<8192, 256, 0, stream>>>(Wx, Uh, Wt);
    pack_fc_w  <<<1024, 256, 0, stream>>>(fcW, fcWt);
    init_state <<< 512, 256, 0, stream>>>(bx, bu, bias, h0, c);
    gather_embed<<<8192, 256, 0, stream>>>(captions, emb, xseq);

    for (int t = 0; t < kT; ++t) {
        const bf16* hi = (t & 1) ? h1 : h0;
        bf16*       ho = (t & 1) ? h0 : h1;
        lstm_step<<<128, 128, 0, stream>>>(xseq + (size_t)t * kB * kE,
                                           hi, ho, Wt, bias, c);
    }
    // 128 steps end with write to h0
    fc_normalize<<<16, 256, 0, stream>>>(h0, fcWt, fcb, (float*)d_out);
}
